// MLPModel_all_12738873000521
// MI455X (gfx1250) — matrix operand order repaired and checked by reading, not compiled, not run on hardware
//
#include <hip/hip_runtime.h>
#include <hip/hip_bf16.h>
#include <math.h>

// ---------------------------------------------------------------------------
// MI455X (gfx1250) implementation: BN-folded MLP pipeline with f16 WMMA GEMMs
//   pack+stats0 -> fold0 -> GEMM1(wmma)+stats1 -> fold1 -> GEMM2(wmma)+stats2
//   -> fold2 -> GEMM3(wmma)+stats3 -> fold3 -> final scalar/interp kernel
// Activations staged in workspace as _Float16; stats/bias/scalar math in fp32.
// Each wave computes 16 rows x (16*NTILES) cols, reusing the A fragment
// across NTILES WMMAs per K-step so activations are read from HBM once.
// ---------------------------------------------------------------------------

typedef _Float16 v16h __attribute__((ext_vector_type(16)));
typedef _Float16 v8h  __attribute__((ext_vector_type(8)));
typedef float    v8f  __attribute__((ext_vector_type(8)));

#define BN_EPS 1e-5f

// ---------------------------------------------------------------------------
// Kernel: gather x0 = x[:, {0..5,7,9}], write f16 padded rows [N][32],
// accumulate per-column sum / sumsq (fp32) for BN0 via wave shuffle reduce.
// ---------------------------------------------------------------------------
__global__ void __launch_bounds__(256)
pack_x0_stats(const float* __restrict__ x, _Float16* __restrict__ x0p,
              float* __restrict__ sum0, float* __restrict__ sq0, int nrows)
{
    int r = blockIdx.x * blockDim.x + threadIdx.x;
    float v[8];
    if (r < nrows) {
        const float* xr = x + (size_t)r * 10;
        v[0] = xr[0]; v[1] = xr[1]; v[2] = xr[2]; v[3] = xr[3];
        v[4] = xr[4]; v[5] = xr[5]; v[6] = xr[7]; v[7] = xr[9];
        _Float16* o = x0p + (size_t)r * 32;
#pragma unroll
        for (int i = 0; i < 8; ++i) o[i] = (_Float16)v[i];
#pragma unroll
        for (int i = 8; i < 32; ++i) o[i] = (_Float16)0.f;
    } else {
#pragma unroll
        for (int i = 0; i < 8; ++i) v[i] = 0.f;
    }
    // Wave-level reduction (wave32) of per-column sum / sumsq.
#pragma unroll
    for (int i = 0; i < 8; ++i) {
        float s  = v[i];
        float s2 = v[i] * v[i];
#pragma unroll
        for (int off = 16; off > 0; off >>= 1) {
            s  += __shfl_xor(s,  off, 32);
            s2 += __shfl_xor(s2, off, 32);
        }
        if ((threadIdx.x & 31) == 0) {
            atomicAdd(&sum0[i], s);
            atomicAdd(&sq0[i],  s2);
        }
    }
}

// ---------------------------------------------------------------------------
// Fold BN0 into layer-1 weights: w1p[256][32] f16 (K padded to 32), b1p[256].
// ---------------------------------------------------------------------------
__global__ void fold0(const float* __restrict__ g0, const float* __restrict__ be0,
                      const float* __restrict__ sum0, const float* __restrict__ sq0,
                      const float* __restrict__ w1, const float* __restrict__ b1,
                      _Float16* __restrict__ w1p, float* __restrict__ b1p, float invN)
{
    __shared__ float s0s[8], sh0s[8];
    if (threadIdx.x < 8) {
        float m   = sum0[threadIdx.x] * invN;
        float var = sq0[threadIdx.x] * invN - m * m;
        float s   = g0[threadIdx.x] * rsqrtf(var + BN_EPS);
        s0s[threadIdx.x]  = s;
        sh0s[threadIdx.x] = be0[threadIdx.x] - m * s;
    }
    __syncthreads();
    int o = threadIdx.x;                 // 256 output features
    float acc = b1[o];
#pragma unroll
    for (int i = 0; i < 8; ++i) {
        float wv = w1[o * 8 + i];
        acc += wv * sh0s[i];
        w1p[o * 32 + i] = (_Float16)(wv * s0s[i]);
    }
#pragma unroll
    for (int i = 8; i < 32; ++i) w1p[o * 32 + i] = (_Float16)0.f;
    b1p[o] = acc;
}

// ---------------------------------------------------------------------------
// Fold BN(stage) into next layer's weights. One block per output row.
//   wp[o][j] = f16(w[o][j] * s[j]);  bp[o] = b[o] + sum_j w[o][j]*shift[j]
// ---------------------------------------------------------------------------
template <int KIN>
__global__ void fold_mid(const float* __restrict__ g, const float* __restrict__ be,
                         const float* __restrict__ sum, const float* __restrict__ sq,
                         const float* __restrict__ w, const float* __restrict__ b,
                         _Float16* __restrict__ wp, float* __restrict__ bp, float invN)
{
    __shared__ float red[KIN];
    int o = blockIdx.x;
    int j = threadIdx.x;
    float m   = sum[j] * invN;
    float var = sq[j] * invN - m * m;
    float s   = g[j] * rsqrtf(var + BN_EPS);
    float sh  = be[j] - m * s;
    float wv  = w[(size_t)o * KIN + j];
    wp[(size_t)o * KIN + j] = (_Float16)(wv * s);
    red[j] = wv * sh;
    __syncthreads();
    for (int st = KIN / 2; st > 0; st >>= 1) {
        if (j < st) red[j] += red[j + st];
        __syncthreads();
    }
    if (j == 0) bp[o] = b[o] + red[0];
}

// ---------------------------------------------------------------------------
// BN3 is applied directly in the final kernel: just compute scale/shift.
// ---------------------------------------------------------------------------
__global__ void fold3(const float* __restrict__ g3, const float* __restrict__ be3,
                      const float* __restrict__ sum3, const float* __restrict__ sq3,
                      float* __restrict__ scale3, float* __restrict__ shift3, float invN)
{
    int j = threadIdx.x;
    if (j < 16) {
        float m   = sum3[j] * invN;
        float var = sq3[j] * invN - m * m;
        float s   = g3[j] * rsqrtf(var + BN_EPS);
        scale3[j] = s;
        shift3[j] = be3[j] - m * s;
    }
}

// ---------------------------------------------------------------------------
// WMMA GEMM + bias + ReLU + per-column stats.
//   A   : [nrows][KTOT] f16 row-major (raw activations, BN folded into W)
//   W   : [NOUT][KTOT]  f16 row-major (= B^T, BN-scaled)
//   out : [nrows][NOUT] f16
// Block = 256 threads = 8 waves. Each wave owns 16 rows x NCOLS cols
// (NCOLS = 16*NTILES tiles), reusing its A fragment across NTILES WMMAs.
// Grid  = (nrows/128, NOUT/NCOLS).
// Epilogue stages the f16 tile in LDS and emits coalesced b128 stores.
// ---------------------------------------------------------------------------
template <int KTOT, int NOUT, int NTILES>
__global__ void __launch_bounds__(256)
gemm_bn_relu_f16(const _Float16* __restrict__ A, const _Float16* __restrict__ W,
                 const float* __restrict__ bias, _Float16* __restrict__ out,
                 float* __restrict__ statSum, float* __restrict__ statSq)
{
    constexpr int NCOLS = 16 * NTILES;

    const int lane = threadIdx.x & 31;
    const int wave = threadIdx.x >> 5;
    const int m16  = lane & 15;
    const int half = lane >> 4;
    const int rowBase = blockIdx.x * 128 + wave * 16;
    const int colBase = blockIdx.y * NCOLS;

    const _Float16* arow = A + (size_t)(rowBase + m16) * KTOT;
    const _Float16* wrow = W + (size_t)(colBase + m16) * KTOT;

    v8f c[NTILES] = {};
#pragma unroll
    for (int kb = 0; kb < KTOT; kb += 32) {
        // A (16x32 f16): lanes 0-15 M=lane; element i -> K=kb+half*16+i
        // (contiguous K per lane half, matching B).
        v8h a0 = *(const v8h*)(arow + kb + half * 16);
        v8h a1 = *(const v8h*)(arow + kb + half * 16 + 8);
        v16h a = __builtin_shufflevector(a0, a1, 0, 1, 2, 3, 4, 5, 6, 7,
                                                 8, 9, 10, 11, 12, 13, 14, 15);
#pragma unroll
        for (int t = 0; t < NTILES; ++t) {
            // B (32x16 f16): lane -> column, 16 contiguous K per lane half.
            v16h b = *(const v16h*)(wrow + (size_t)t * 16 * KTOT + kb + half * 16);
            c[t] = __builtin_amdgcn_wmma_f32_16x16x32_f16(
                       /*neg_a=*/false, a, /*neg_b=*/false, b,
                       /*c_mod=*/(short)0, c[t], /*reuse_a=*/false, /*reuse_b=*/false);
        }
    }

    // Epilogue: bias + ReLU + stats; stage f16 tile in LDS for b128 stores.
    __shared__ _Float16 stg[8 * 16 * NCOLS];
    __shared__ float ssum[NCOLS], ssq[NCOLS];
    if (threadIdx.x < NCOLS) { ssum[threadIdx.x] = 0.f; ssq[threadIdx.x] = 0.f; }
    __syncthreads();

    _Float16* wstg = stg + wave * 16 * NCOLS;
#pragma unroll
    for (int t = 0; t < NTILES; ++t) {
        const float bv = bias[colBase + t * 16 + m16];
        float s = 0.f, s2 = 0.f;
#pragma unroll
        for (int r = 0; r < 8; ++r) {
            float v = c[t][r] + bv;
            v = v > 0.f ? v : 0.f;
            wstg[(8 * half + r) * NCOLS + t * 16 + m16] = (_Float16)v;
            s += v; s2 += v * v;
        }
        atomicAdd(&ssum[t * 16 + m16], s);
        atomicAdd(&ssq [t * 16 + m16], s2);
    }
    __syncthreads();

    // Coalesced stores: 16 rows x NCOLS f16 = 2*NCOLS 16B chunks per wave.
    constexpr int CHUNKS_PER_ROW = NCOLS / 8;      // 16B chunks per row
#pragma unroll
    for (int i = 0; i < NTILES; ++i) {             // 2*NCOLS/32 iterations
        int e = i * 32 + lane;
        int row = e / CHUNKS_PER_ROW;
        int cc  = e % CHUNKS_PER_ROW;
        v8h val = *(const v8h*)(wstg + row * NCOLS + cc * 8);
        *(v8h*)(out + (size_t)(rowBase + row) * NOUT + colBase + cc * 8) = val;
    }
    if (threadIdx.x < NCOLS) {
        atomicAdd(&statSum[colBase + threadIdx.x], ssum[threadIdx.x]);
        atomicAdd(&statSq [colBase + threadIdx.x], ssq [threadIdx.x]);
    }
}

// ---------------------------------------------------------------------------
// Bilinear interpolation helpers (grids are tiny; scan = searchsorted 'right').
// ---------------------------------------------------------------------------
__device__ __forceinline__ int grid_bucket(const float* __restrict__ g, int n, float q)
{
    int c = 0;
    for (int j = 1; j < n; ++j) c += (g[j] <= q) ? 1 : 0;
    return c < (n - 2) ? c : (n - 2);
}

__device__ __forceinline__ float interp2d(const float* __restrict__ xg, int nx,
                                          const float* __restrict__ yg, int ny,
                                          const float* __restrict__ tab,
                                          float xq, float yq)
{
    int xi = grid_bucket(xg, nx, xq);
    int yi = grid_bucket(yg, ny, yq);
    float tx = (xq - xg[xi]) / (xg[xi + 1] - xg[xi]);
    float ty = (yq - yg[yi]) / (yg[yi + 1] - yg[yi]);
    tx = fminf(fmaxf(tx, 0.f), 1.f);
    ty = fminf(fmaxf(ty, 0.f), 1.f);
    float f00 = tab[xi * ny + yi];
    float f01 = tab[xi * ny + yi + 1];
    float f10 = tab[(xi + 1) * ny + yi];
    float f11 = tab[(xi + 1) * ny + yi + 1];
    return (1.f - tx) * ((1.f - ty) * f00 + ty * f01)
         +        tx  * ((1.f - ty) * f10 + ty * f11);
}

// ---------------------------------------------------------------------------
// Final per-row kernel: BN3 + tiny 16->3 layer + interpolation pipeline.
// ---------------------------------------------------------------------------
__global__ void __launch_bounds__(256)
final_kernel(const float* __restrict__ x, const _Float16* __restrict__ a3,
             const float* __restrict__ scale3, const float* __restrict__ shift3,
             const float* __restrict__ w4, const float* __restrict__ b4,
             const float* __restrict__ kptr,
             const float* __restrict__ rpm_grid, const float* __restrict__ pr_grid,
             const float* __restrict__ ie_table,
             const float* __restrict__ plp_grid, const float* __restrict__ ph_grid,
             const float* __restrict__ pt_table,
             const float* __restrict__ cht_grid, const float* __restrict__ chp_grid,
             const float* __restrict__ chh_table,
             float* __restrict__ out, int nrows)
{
    int r = blockIdx.x * blockDim.x + threadIdx.x;
    if (r >= nrows) return;

    float hb[16];
    const _Float16* ar = a3 + (size_t)r * 16;
#pragma unroll
    for (int j = 0; j < 16; ++j) hb[j] = (float)ar[j] * scale3[j] + shift3[j];

    float x8[3];
#pragma unroll
    for (int c = 0; c < 3; ++c) {
        float acc = b4[c];
#pragma unroll
        for (int j = 0; j < 16; ++j) acc += w4[c * 16 + j] * hb[j];
        x8[c] = acc;
    }

    const float* xr = x + (size_t)r * 10;
    float speed = xr[2], lo_p = xr[5], t1 = xr[4];
    float cc = x8[1];
    float hi_p = fmaxf(x8[2], lo_p);
    float pr = hi_p / lo_p;

    float fie = interp2d(rpm_grid, 11, pr_grid, 6, ie_table, speed, pr);
    float h1v = interp2d(cht_grid, 64, chp_grid, 64, chh_table, t1, lo_p);
    float kk = kptr[0];
    float t2 = (t1 + 273.15f) * expf(kk * logf(pr)) - 273.15f;
    float h2r = interp2d(cht_grid, 64, chp_grid, 64, chh_table, t2, hi_p);
    float h2t = (h2r - h1v) / fie + h1v;
    float th2 = interp2d(plp_grid, 64, ph_grid, 128, pt_table, logf(hi_p), h2t)
              + 0.1f * cc;

    out[(size_t)r * 3 + 0] = x8[0];
    out[(size_t)r * 3 + 1] = th2;
    out[(size_t)r * 3 + 2] = x8[2];
}

// ---------------------------------------------------------------------------
// Host orchestration (graph-capture safe: only async ops on `stream`).
// ---------------------------------------------------------------------------
extern "C" void kernel_launch(void* const* d_in, const int* in_sizes, int n_in,
                              void* d_out, int out_size, void* d_ws, size_t ws_size,
                              hipStream_t stream)
{
    const float* x   = (const float*)d_in[0];
    const float* w1  = (const float*)d_in[1];
    const float* b1  = (const float*)d_in[2];
    const float* w2  = (const float*)d_in[3];
    const float* b2  = (const float*)d_in[4];
    const float* w3  = (const float*)d_in[5];
    const float* b3  = (const float*)d_in[6];
    const float* w4  = (const float*)d_in[7];
    const float* b4  = (const float*)d_in[8];
    const float* g0  = (const float*)d_in[9];
    const float* be0 = (const float*)d_in[10];
    const float* g1  = (const float*)d_in[11];
    const float* be1 = (const float*)d_in[12];
    const float* g2  = (const float*)d_in[13];
    const float* be2 = (const float*)d_in[14];
    const float* g3  = (const float*)d_in[15];
    const float* be3 = (const float*)d_in[16];
    const float* kp  = (const float*)d_in[17];
    const float* rpm_grid = (const float*)d_in[18];
    const float* pr_grid  = (const float*)d_in[19];
    const float* ie_table = (const float*)d_in[20];
    const float* plp_grid = (const float*)d_in[21];
    const float* ph_grid  = (const float*)d_in[22];
    const float* pt_table = (const float*)d_in[23];
    const float* cht_grid = (const float*)d_in[24];
    const float* chp_grid = (const float*)d_in[25];
    const float* chh_table= (const float*)d_in[26];
    (void)n_in; (void)out_size; (void)ws_size;

    const int nrows = in_sizes[0] / 10;           // 262144
    const float invN = 1.0f / (float)nrows;

    // Workspace carve-out (256B aligned chunks).
    char* ws = (char*)d_ws;
    size_t off = 0;
    auto carve = [&](size_t bytes) -> void* {
        void* p = ws + off;
        off = (off + bytes + 255) & ~(size_t)255;
        return p;
    };
    _Float16* x0p  = (_Float16*)carve((size_t)nrows * 32 * sizeof(_Float16));   // 16 MiB
    _Float16* h1b  = (_Float16*)carve((size_t)nrows * 256 * sizeof(_Float16));  // 128 MiB
    _Float16* h2b  = (_Float16*)carve((size_t)nrows * 128 * sizeof(_Float16));  // 64 MiB
    _Float16* a3b  = (_Float16*)carve((size_t)nrows * 16 * sizeof(_Float16));   // 8 MiB
    _Float16* w1p  = (_Float16*)carve(256 * 32 * sizeof(_Float16));
    _Float16* w2p  = (_Float16*)carve(128 * 256 * sizeof(_Float16));
    _Float16* w3p  = (_Float16*)carve(16 * 128 * sizeof(_Float16));
    float*    b1p  = (float*)carve(256 * sizeof(float));
    float*    b2p  = (float*)carve(128 * sizeof(float));
    float*    b3p  = (float*)carve(16 * sizeof(float));
    float*    stats= (float*)carve(816 * sizeof(float));
    float*    scale3 = (float*)carve(16 * sizeof(float));
    float*    shift3 = (float*)carve(16 * sizeof(float));

    float* sum0 = stats;        float* sq0 = stats + 8;
    float* sum1 = stats + 16;   float* sq1 = stats + 272;
    float* sum2 = stats + 528;  float* sq2 = stats + 656;
    float* sum3 = stats + 784;  float* sq3 = stats + 800;

    // Zero the stats accumulators (graph-capturable async memset).
    hipMemsetAsync(stats, 0, 816 * sizeof(float), stream);

    // Stage 0: pack x0 to f16 + BN0 stats.
    pack_x0_stats<<<(nrows + 255) / 256, 256, 0, stream>>>(x, x0p, sum0, sq0, nrows);
    fold0<<<1, 256, 0, stream>>>(g0, be0, sum0, sq0, w1, b1, w1p, b1p, invN);

    // Stage 1: [N,8(->32)] x [32,256] WMMA, ReLU, stats for BN1.
    gemm_bn_relu_f16<32, 256, 8><<<dim3(nrows / 128, 2), 256, 0, stream>>>(
        x0p, w1p, b1p, h1b, sum1, sq1);
    fold_mid<256><<<128, 256, 0, stream>>>(g1, be1, sum1, sq1, w2, b2, w2p, b2p, invN);

    // Stage 2: [N,256] x [256,128] WMMA (dominant GEMM), ReLU, stats for BN2.
    // One block covers all 128 output columns -> h1 is read from HBM once.
    gemm_bn_relu_f16<256, 128, 8><<<dim3(nrows / 128, 1), 256, 0, stream>>>(
        h1b, w2p, b2p, h2b, sum2, sq2);
    fold_mid<128><<<16, 128, 0, stream>>>(g2, be2, sum2, sq2, w3, b3, w3p, b3p, invN);

    // Stage 3: [N,128] x [128,16] WMMA, ReLU, stats for BN3.
    gemm_bn_relu_f16<128, 16, 1><<<dim3(nrows / 128, 1), 256, 0, stream>>>(
        h2b, w3p, b3p, a3b, sum3, sq3);
    fold3<<<1, 16, 0, stream>>>(g3, be3, sum3, sq3, scale3, shift3, invN);

    // Stage 4: BN3 + 16->3 layer + interpolation pipeline (fp32).
    final_kernel<<<(nrows + 255) / 256, 256, 0, stream>>>(
        x, a3b, scale3, shift3, w4, b4, kp,
        rpm_grid, pr_grid, ie_table, plp_grid, ph_grid, pt_table,
        cht_grid, chp_grid, chh_table, (float*)d_out, nrows);
}
